// LateralLayer_89455578841259
// MI455X (gfx1250) — matrix operand order repaired and checked by reading, not compiled, not run on hardware
//
#include <hip/hip_runtime.h>
#include <hip/hip_bf16.h>
#include <stdint.h>

typedef __attribute__((ext_vector_type(16))) _Float16 v16h;
typedef __attribute__((ext_vector_type(8)))  float    v8f;

#define BB   4
#define CIN  64
#define COUT 64
#define HH   128
#define WW   128
#define KK   5
#define TAPS 25
#define HW   (HH * WW)          // 16384
#define NEL  (BB * COUT * HW)   // 4194304
#define WCOL 132                // 128 + 2*2 pad columns
#define HROW 132                // 128 + 2*2 pad rows
#define LDSH (KK * WCOL * CIN)  // 42240 halves = 84480 bytes
#define SLAB_BYTES (LDSH * 2)   // 84480
#define SLAB_V16 (SLAB_BYTES / 16)  // 5280 16-byte units

// ---------------------------------------------------------------------------
// W (64 x 1600 f32) -> f16 WMMA A-fragments, per-lane contiguous-K layout (A 16x32):
//   lanes 0-15 : halves 0..15 -> K=0..15
//   lanes 16-31: halves 0..15 -> K=16..31
// frag = ((t*2 + cc)*4 + mt), 512 halves each.
// ---------------------------------------------------------------------------
__global__ void k_prepw(const float* __restrict__ W, _Float16* __restrict__ Wf) {
    int idx = blockIdx.x * blockDim.x + threadIdx.x;      // 0 .. 102399
    int frag = idx >> 9;
    int within = idx & 511;
    int lane = within >> 4;
    int half = within & 15;
    int t  = frag >> 3;
    int cc = (frag >> 2) & 1;
    int mt = frag & 3;
    int M = mt * 16 + (lane & 15);
    int base = half;
    int K = base + ((lane >= 16) ? 16 : 0);
    int c = cc * 32 + K;
    Wf[idx] = (_Float16)W[M * (CIN * TAPS) + c * TAPS + t];
}

__global__ void k_wsum(const float* __restrict__ W, float* __restrict__ wsum) {
    int o = threadIdx.x;
    if (o < COUT) {
        float s = 0.0f;
        for (int k = 0; k < CIN * TAPS; ++k) s += W[o * (CIN * TAPS) + k];
        wsum[o] = s;
    }
}

// ---------------------------------------------------------------------------
// x (f32, NCHW) -> xh (f16) in padded transposed layout [b][hp 0..131][wp
// 0..131][c], zero borders; also per-pixel channel sum (colsum).
// ---------------------------------------------------------------------------
__global__ void k_prepx(const float* __restrict__ x,
                        _Float16* __restrict__ xh,
                        float* __restrict__ colsum) {
    int idx = blockIdx.x * blockDim.x + threadIdx.x;      // over BB*132*132
    if (idx >= BB * HROW * WCOL) return;
    int wp = idx % WCOL;
    int hp = (idx / WCOL) % HROW;
    int b  = idx / (WCOL * HROW);
    int h = hp - 2;
    int w = wp - 2;
    bool inb = (h >= 0 && h < HH && w >= 0 && w < WW);

    _Float16 tmp[CIN];
    float s = 0.0f;
#pragma unroll
    for (int c = 0; c < CIN; ++c) {
        float v = inb ? x[(size_t)(b * CIN + c) * HW + h * WW + w] : 0.0f;
        s += v;
        tmp[c] = (_Float16)v;
    }
    _Float16* dst = xh + ((size_t)(b * HROW + hp) * WCOL + wp) * CIN;
#pragma unroll
    for (int q = 0; q < 4; ++q)
        *(v16h*)(dst + q * 16) = *(const v16h*)(tmp + q * 16);
    if (inb) colsum[b * HW + h * WW + w] = s;
}

// 5x5 window sum of colsum (zero pad), clamped at K=5
__global__ void k_winsum(const float* __restrict__ colsum, float* __restrict__ wact) {
    int idx = blockIdx.x * blockDim.x + threadIdx.x;      // 0 .. 65535
    int b = idx >> 14;
    int hw = idx & (HW - 1);
    int h = hw >> 7;
    int w = hw & (WW - 1);
    float s = 0.0f;
    for (int dj = 0; dj < KK; ++dj) {
        int hh = h + dj - 2;
        if (hh < 0 || hh >= HH) continue;
        for (int dk = 0; dk < KK; ++dk) {
            int ww = w + dk - 2;
            if (ww < 0 || ww >= WW) continue;
            s += colsum[(b * HH + hh) * WW + ww];
        }
    }
    wact[idx] = fmaxf(s, (float)KK);
}

// ---------------------------------------------------------------------------
// Implicit-GEMM conv via WMMA. One workgroup per (b, h); the 5-row f16 slab
// is contiguous in xh -> fill is pure GLOBAL_LOAD_ASYNC_TO_LDS_B128 +
// s_wait_asynccnt. B-fragment reads: dj=0..3 via immediate offsets from pB0
// (max 51200 B); dj=4 via an opaque base so no >64KB immediates are needed.
// ---------------------------------------------------------------------------
__global__ void k_conv(const _Float16* __restrict__ xh,
                       const _Float16* __restrict__ Wf,
                       float* __restrict__ xl) {
    extern __shared__ _Float16 lds[];
    int b = blockIdx.x / HH;
    int h = blockIdx.x % HH;
    int tid = threadIdx.x;
    int lane = tid & 31;
    int wave = tid >> 5;

    // Async-copy the contiguous 84480-byte slab (rows hp=h..h+4).
    {
        const char* src = (const char*)(xh + (size_t)(b * HROW + h) * WCOL * CIN);
        unsigned lds0 = (unsigned)(uintptr_t)lds;   // low 32 bits = LDS offset
        for (int u = tid; u < SLAB_V16; u += 256) {
            unsigned loff = lds0 + (unsigned)(u * 16);
            const void* g = src + (size_t)u * 16;
            asm volatile("global_load_async_to_lds_b128 %0, %1, off"
                         :: "v"(loff), "v"(g) : "memory");
        }
        asm volatile("s_wait_asynccnt 0" ::: "memory");
    }
    __syncthreads();

    v8f acc0 = {}, acc1 = {}, acc2 = {}, acc3 = {};
    int w0 = wave * 16;
    int npix = lane & 15;
    int cb_hi = (lane >> 4) << 4;   // 0 or 16

    const _Float16* pB0 = lds + (w0 + npix) * CIN + cb_hi;
    int off4 = 4 * WCOL * CIN;                 // dj=4 row, in halves
    asm volatile("" : "+v"(off4));             // opaque: block const-folding
    const _Float16* pB4 = pB0 + off4;          // still provably LDS
    const _Float16* aw  = Wf + lane * 16;

#pragma unroll
    for (int t = 0; t < TAPS; ++t) {
        const int dj = t / KK;
        const int dk = t - dj * KK;
        const _Float16* pb = (dj < 4) ? pB0 : pB4;
        const int roff = (((dj < 4) ? dj : 0) * WCOL + dk) * CIN;
        __builtin_prefetch(aw + (t * 2 + 2) * 4 * 512, 0, 3);
#pragma unroll
        for (int cc = 0; cc < 2; ++cc) {
            v16h bfrag = *(const v16h*)(pb + roff + cc * 32);
            const _Float16* wf = aw + (size_t)((t * 2 + cc) * 4) * 512;
            v16h a0 = *(const v16h*)(wf);
            v16h a1 = *(const v16h*)(wf + 512);
            v16h a2 = *(const v16h*)(wf + 1024);
            v16h a3 = *(const v16h*)(wf + 1536);
            acc0 = __builtin_amdgcn_wmma_f32_16x16x32_f16(false, a0, false, bfrag, (short)0, acc0, false, false);
            acc1 = __builtin_amdgcn_wmma_f32_16x16x32_f16(false, a1, false, bfrag, (short)0, acc1, false, false);
            acc2 = __builtin_amdgcn_wmma_f32_16x16x32_f16(false, a2, false, bfrag, (short)0, acc2, false, false);
            acc3 = __builtin_amdgcn_wmma_f32_16x16x32_f16(false, a3, false, bfrag, (short)0, acc3, false, false);
        }
    }

    // C/D layout: VGPR r: lanes 0-15 -> M=r, lanes 16-31 -> M=8+r; N = lane&15.
    int mbase = (lane >> 4) * 8;
    size_t px = (size_t)h * WW + w0 + npix;
#pragma unroll
    for (int r = 0; r < 8; ++r) {
        int m = mbase + r;
        xl[((size_t)(b * COUT + 0  + m) * HW) + px] = acc0[r];
        xl[((size_t)(b * COUT + 16 + m) * HW) + px] = acc1[r];
        xl[((size_t)(b * COUT + 32 + m) * HW) + px] = acc2[r];
        xl[((size_t)(b * COUT + 48 + m) * HW) + px] = acc3[r];
    }
}

// per-(b,o) spatial max of xl / wact / (1e-10 + wsum[o])
__global__ void k_rowmax(const float* __restrict__ xl,
                         const float* __restrict__ wact,
                         const float* __restrict__ wsum,
                         float* __restrict__ gmax) {
    __shared__ float red[256];
    int bo = blockIdx.x;            // 0..255
    int b = bo >> 6;
    int o = bo & 63;
    int tid = threadIdx.x;
    float invw = 1.0f / (1e-10f + wsum[o]);
    float m = -3.0e38f;
    for (int hw = tid; hw < HW; hw += 256) {
        float v = xl[(size_t)bo * HW + hw] / wact[b * HW + hw] * invw;
        m = fmaxf(m, v);
    }
    red[tid] = m;
    __syncthreads();
    for (int s = 128; s > 0; s >>= 1) {
        if (tid < s) red[tid] = fmaxf(red[tid], red[tid + s]);
        __syncthreads();
    }
    if (tid == 0) gmax[bo] = red[0];
}

__global__ void k_final(const float* __restrict__ xl,
                        const float* __restrict__ wact,
                        const float* __restrict__ wsum,
                        const float* __restrict__ gmax,
                        float* __restrict__ xn,
                        float* __restrict__ xbin) {
    int idx = blockIdx.x * blockDim.x + threadIdx.x;      // 0 .. NEL-1
    int bo = idx >> 14;
    int b = bo >> 6;
    int o = bo & 63;
    int hw = idx & (HW - 1);
    float v = xl[idx] / wact[b * HW + hw] / (1e-10f + wsum[o]);
    float n = v / (1e-10f + gmax[bo]);
    xn[idx] = n;
    float c3 = n * n * n;
    xbin[idx] = (c3 >= 0.5f) ? 1.0f : 0.0f;
}

extern "C" void kernel_launch(void* const* d_in, const int* in_sizes, int n_in,
                              void* d_out, int out_size, void* d_ws, size_t ws_size,
                              hipStream_t stream) {
    (void)in_sizes; (void)n_in; (void)out_size; (void)ws_size;
    const float* x = (const float*)d_in[0];
    const float* W = (const float*)d_in[1];
    float* out = (float*)d_out;

    char* ws = (char*)d_ws;
    _Float16* Wf   = (_Float16*)ws;               // 204800 B (102400 halves)
    float* wsum    = (float*)(ws + 204800);       // 256 B
    float* colsum  = (float*)(ws + 205056);       // 262144 B
    float* wact    = (float*)(ws + 467200);       // 262144 B
    float* gmax    = (float*)(ws + 729344);       // 1024 B
    _Float16* xh   = (_Float16*)(ws + 730368);    // 4*132*132*64*2 = 8921088 B

    float* xl   = out;
    float* xn   = out + (size_t)NEL;
    float* xbin = out + (size_t)(2 * NEL);

    k_prepw <<<400, 256, 0, stream>>>(W, Wf);
    k_wsum  <<<1,    64, 0, stream>>>(W, wsum);
    k_prepx <<<(BB * HROW * WCOL + 255) / 256, 256, 0, stream>>>(x, xh, colsum);
    k_winsum<<<256, 256, 0, stream>>>(colsum, wact);
    k_conv  <<<BB * HH, 256, SLAB_BYTES, stream>>>(xh, Wf, xl);
    k_rowmax<<<256, 256, 0, stream>>>(xl, wact, wsum, gmax);
    k_final <<<NEL / 256, 256, 0, stream>>>(xl, wact, wsum, gmax, xn, xbin);
}
